// Model_1116691497184
// MI455X (gfx1250) — matrix operand order repaired and checked by reading, not compiled, not run on hardware
//
#include <hip/hip_runtime.h>
#include <hip/hip_bf16.h>
#include <math.h>

#define B_      8
#define L_      2048
#define DMODEL  1024
#define DINNER  2048
#define DSTATE  16
#define DTRANK  64
#define XZS     (2*DINNER)           // 4096
#define M_ROWS  (B_*L_)              // 16384
#define EPS_    1e-5f

typedef __bf16 bf16_t;
typedef bf16_t v16bf __attribute__((ext_vector_type(16)));
typedef float  v8f   __attribute__((ext_vector_type(8)));
typedef int    vs4i  __attribute__((vector_size(16)));   // matches builtin param

union FragU { v16bf v; uint4 u[2]; };

// ---- CDNA5 async global->LDS copy (ASYNCcnt path), with safe fallback -----
#if defined(__has_builtin)
#if __has_builtin(__builtin_amdgcn_global_load_async_to_lds_b128)
#define HAVE_ASYNC_LDS 1
#endif
#endif
#ifndef HAVE_ASYNC_LDS
#define HAVE_ASYNC_LDS 0
#endif

#define GAS1 __attribute__((address_space(1)))
#define LAS3 __attribute__((address_space(3)))

__device__ __forceinline__ void async_copy16(const void* g, void* l) {
#if HAVE_ASYNC_LDS
  __builtin_amdgcn_global_load_async_to_lds_b128(
      (GAS1 vs4i*)(vs4i*)g, (LAS3 vs4i*)(vs4i*)l, 0, 0);
#else
  *(uint4*)l = *(const uint4*)g;
#endif
}

__device__ __forceinline__ void wait_async_zero() {
#if HAVE_ASYNC_LDS
#if __has_builtin(__builtin_amdgcn_s_wait_asynccnt)
  __builtin_amdgcn_s_wait_asynccnt(0);
#else
  asm volatile("s_wait_asynccnt 0" ::: "memory");
#endif
#endif
}

// ---------------------------------------------------------------------------
// f32 -> bf16 elementwise converter (weights / GEMM operand views)
// ---------------------------------------------------------------------------
__global__ __launch_bounds__(256) void cvt_f32_bf16_kernel(
    const float* __restrict__ s, bf16_t* __restrict__ d, int count4)
{
  const int idx = blockIdx.x * 256 + threadIdx.x;
  if (idx < count4) {
    float4 f = ((const float4*)s)[idx];
    d[idx * 4 + 0] = (bf16_t)f.x; d[idx * 4 + 1] = (bf16_t)f.y;
    d[idx * 4 + 2] = (bf16_t)f.z; d[idx * 4 + 3] = (bf16_t)f.w;
  }
}

// ---------------------------------------------------------------------------
// 1) fused residual-add + LayerNorm + position embedding
//    one 256-thread block per row of 1024; h written directly as bf16
// ---------------------------------------------------------------------------
__global__ __launch_bounds__(256) void ln_fused_kernel(
    const float* __restrict__ hidden, const float* __restrict__ resid_in,
    const float* __restrict__ pos,    const float* __restrict__ w,
    const float* __restrict__ bia,
    float* __restrict__ resid_out, bf16_t* __restrict__ h_out)
{
  const int row = blockIdx.x;
  const size_t base = (size_t)row * DMODEL;
  const int tid = threadIdx.x;
  float vals[4];
  float s = 0.f, q = 0.f;
#pragma unroll
  for (int i = 0; i < 4; ++i) {
    int c = tid + i * 256;
    float v = hidden[base + c] + resid_in[base + c];
    vals[i] = v;
    resid_out[base + c] = v;
    s += v; q += v * v;
  }
#pragma unroll
  for (int off = 16; off > 0; off >>= 1) {
    s += __shfl_xor(s, off, 32);
    q += __shfl_xor(q, off, 32);
  }
  __shared__ float ss[8], sq[8];
  if ((tid & 31) == 0) { ss[tid >> 5] = s; sq[tid >> 5] = q; }
  __syncthreads();
  float ts = 0.f, tq = 0.f;
#pragma unroll
  for (int i = 0; i < 8; ++i) { ts += ss[i]; tq += sq[i]; }
  const float mu   = ts * (1.f / DMODEL);
  const float var  = tq * (1.f / DMODEL) - mu * mu;
  const float rstd = rsqrtf(var + EPS_);
#pragma unroll
  for (int i = 0; i < 4; ++i) {
    int c = tid + i * 256;
    h_out[base + c] = (bf16_t)((vals[i] - mu) * rstd * w[c] + bia[c] + pos[base + c]);
  }
}

// ---------------------------------------------------------------------------
// 2) WMMA bf16 GEMM:  C[m,n] = sum_k A[m,k] * W[n,k]   (A, W pre-cast bf16)
//    Tile 128x128x32, 8 waves (4Mx2N), each wave 2x4 16x16 tiles -> 8 WMMA/step.
//    Double-buffered LDS filled via global_load_async_to_lds_b128 (ASYNCcnt),
//    overlapping the k+1 tile copy with the current tile's WMMAs.
//    EPI==1: fused  softplus(x + bias[n])  (dt_proj epilogue)
// ---------------------------------------------------------------------------
template<int EPI>
__global__ __launch_bounds__(256) void gemm_wmma_bf16(
    const bf16_t* __restrict__ A, int lda,
    const bf16_t* __restrict__ W, int K,
    float* __restrict__ C, int ldc, int N,
    const float* __restrict__ bias)
{
  __shared__ bf16_t sA[2][128 * 40];   // [m][k], stride 40 bf16 (80B, 16B align)
  __shared__ bf16_t sB[2][128 * 40];   // [n][k]

  const int tid  = threadIdx.x;
  const int lane = tid & 31;
  const int wid  = tid >> 5;
  const int wm   = wid & 3;          // wave M coord (0..3)
  const int wn   = wid >> 2;         // wave N coord (0..1)
  const int lr   = lane & 15;
  const int lh   = lane >> 4;

  const int m0 = blockIdx.y * 128;
  const int n0 = blockIdx.x * 128;

  v8f acc[2][4] = {};

  // cooperative tile copy: 512 16B chunks per tile, 2 per thread
  auto load_tiles = [&](int k0, int buf) {
#pragma unroll
    for (int c = 0; c < 2; ++c) {
      const int cid = tid * 2 + c;         // 0..511
      const int r = cid >> 2;              // tile row 0..127
      const int q = cid & 3;               // 16B chunk within 32-elem row
      async_copy16(A + (size_t)(m0 + r) * lda + k0 + q * 8,
                   &sA[buf][r * 40 + q * 8]);
      const int n = n0 + r;
      if (n < N) {
        async_copy16(W + (size_t)n * K + k0 + q * 8,
                     &sB[buf][r * 40 + q * 8]);
      } else {
        *(uint4*)&sB[buf][r * 40 + q * 8] = make_uint4(0, 0, 0, 0);
      }
    }
  };

  load_tiles(0, 0);
  int ping = 0;

  for (int k0 = 0; k0 < K; k0 += 32) {
    wait_async_zero();       // current tile landed in LDS
    __syncthreads();         // visible to all waves; prev reads retired
    if (k0 + 32 < K) load_tiles(k0 + 32, ping ^ 1);   // overlap with compute

    // ---- fragments (ISA 7.12.2 layouts) ----
    FragU a[2], b[4];
#pragma unroll
    for (int i = 0; i < 2; ++i) {
      const int m = wm * 32 + i * 16 + lr;
      // elements 0..7: K = lh*8+j ; elements 8..15: K = 16+lh*8+j
      a[i].u[0] = *(const uint4*)&sA[ping][m * 40 + lh * 8];
      a[i].u[1] = *(const uint4*)&sA[ping][m * 40 + 16 + lh * 8];
    }
#pragma unroll
    for (int j = 0; j < 4; ++j) {
      const int n = wn * 64 + j * 16 + lr;
      // elements 0..7: K = lh*8+j ; elements 8..15: K = 16+lh*8+j
      b[j].u[0] = *(const uint4*)&sB[ping][n * 40 + lh * 8];
      b[j].u[1] = *(const uint4*)&sB[ping][n * 40 + 16 + lh * 8];
    }
#pragma unroll
    for (int i = 0; i < 2; ++i)
#pragma unroll
      for (int j = 0; j < 4; ++j)
        acc[i][j] = __builtin_amdgcn_wmma_f32_16x16x32_bf16(
            false, a[i].v, false, b[j].v, (short)0, acc[i][j], false, false);

    ping ^= 1;
  }

  // ---- epilogue:  C/D layout: lane<16 -> M=v ; lane>=16 -> M=8+v ; N=lane%16
#pragma unroll
  for (int i = 0; i < 2; ++i) {
#pragma unroll
    for (int j = 0; j < 4; ++j) {
      const int n = n0 + wn * 64 + j * 16 + lr;
      if (n < N) {
#pragma unroll
        for (int v = 0; v < 8; ++v) {
          const int m = m0 + wm * 32 + i * 16 + lh * 8 + v;
          float val = acc[i][j][v];
          if (EPI == 1) {
            val += bias[n];
            val = (val > 20.f) ? val : log1pf(__expf(val));  // softplus
          }
          C[(size_t)m * ldc + n] = val;
        }
      }
    }
  }
}

// ---------------------------------------------------------------------------
// 3) causal depthwise conv1d (K=4) + SiLU. FLIP=1 reads reversed sequence and
//    materializes activations in flipped coordinates. Dual f32/bf16 outputs.
// ---------------------------------------------------------------------------
template<int FLIP>
__global__ __launch_bounds__(256) void conv_silu_kernel(
    const float* __restrict__ x /* stride XZS */,
    const float* __restrict__ cw, const float* __restrict__ cb,
    float* __restrict__ xact_f, bf16_t* __restrict__ xact_h)
{
  const long long idx = (long long)blockIdx.x * 256 + threadIdx.x;
  const int d = (int)(idx & (DINNER - 1));
  const int l = (int)((idx >> 11) & (L_ - 1));
  const int b = (int)(idx >> 22);
  float acc = cb[d];
#pragma unroll
  for (int j = 0; j < 4; ++j) {
    const int lp = l - 3 + j;
    if (lp >= 0) {
      const int ls = FLIP ? (L_ - 1 - lp) : lp;
      acc += cw[d * 4 + j] * x[((size_t)b * L_ + ls) * XZS + d];
    }
  }
  acc = acc * (1.f / (1.f + __expf(-acc)));                    // SiLU
  const size_t o = ((size_t)b * L_ + l) * DINNER + d;
  xact_f[o] = acc;
  xact_h[o] = (bf16_t)acc;
}

// ---------------------------------------------------------------------------
// 4) selective scan. One thread per (b,d,n); 16-lane shuffle reduction over n.
//    Fuses D-skip term, SiLU(z) gate and (for FLIP) flip-accumulate.
// ---------------------------------------------------------------------------
template<int FLIP>
__global__ __launch_bounds__(256) void scan_kernel(
    const float* __restrict__ xact, const float* __restrict__ dt,
    const float* __restrict__ xdbl /* stride 96; B at +64, C at +80 */,
    const float* __restrict__ A_log, const float* __restrict__ Dp,
    const float* __restrict__ z /* stride XZS */,
    float* __restrict__ ycomb)
{
  const int idx = blockIdx.x * 256 + threadIdx.x;
  const int n  = idx & 15;
  const int bd = idx >> 4;
  const int d  = bd & (DINNER - 1);
  const int b  = bd >> 11;

  const float Ac = -__expf(A_log[d * DSTATE + n]);
  const float Dc = Dp[d];
  float h = 0.f;
  const size_t rowbase = (size_t)b * L_;

  for (int l = 0; l < L_; ++l) {
    const size_t r = rowbase + l;
    const float xv  = xact[r * DINNER + d];
    const float dtv = dt[r * DINNER + d];
    const float Bv  = xdbl[r * 96 + 64 + n];
    const float Cv  = xdbl[r * 96 + 80 + n];
    const float dA  = __expf(dtv * Ac);
    h = dA * h + (dtv * xv) * Bv;
    float c = h * Cv;
#pragma unroll
    for (int off = 8; off > 0; off >>= 1) c += __shfl_xor(c, off, 16);
    if (n == 0) {
      const int lo = FLIP ? (L_ - 1 - l) : l;
      const float zl = z[(rowbase + lo) * XZS + d];
      const float gate = zl * (1.f / (1.f + __expf(-zl)));
      const float out = (c + xv * Dc) * gate;
      const size_t o = (rowbase + lo) * DINNER + d;
      if (FLIP) ycomb[o] += out; else ycomb[o] = out;
    }
  }
}

// ---------------------------------------------------------------------------
extern "C" void kernel_launch(void* const* d_in, const int* in_sizes, int n_in,
                              void* d_out, int out_size, void* d_ws, size_t ws_size,
                              hipStream_t stream)
{
  const float* hidden = (const float*)d_in[0];
  const float* resid  = (const float*)d_in[1];
  const float* pos    = (const float*)d_in[2];
  const float* nw     = (const float*)d_in[3];
  const float* nb     = (const float*)d_in[4];
  const float* in_w   = (const float*)d_in[5];
  const float* out_w  = (const float*)d_in[6];
  const float* conv_w[2] = {(const float*)d_in[7],  (const float*)d_in[14]};
  const float* conv_b[2] = {(const float*)d_in[8],  (const float*)d_in[15]};
  const float* xprj[2]   = {(const float*)d_in[9],  (const float*)d_in[16]};
  const float* dtw[2]    = {(const float*)d_in[10], (const float*)d_in[17]};
  const float* dtb[2]    = {(const float*)d_in[11], (const float*)d_in[18]};
  const float* alog[2]   = {(const float*)d_in[12], (const float*)d_in[19]};
  const float* Dvec[2]   = {(const float*)d_in[13], (const float*)d_in[20]};

  float* out       = (float*)d_out;                         // (B,L,DMODEL)
  float* resid_out = out + (size_t)M_ROWS * DMODEL;         // (B,L,DMODEL)

  // ---- workspace layout: f32 region then bf16 region (all 16B aligned) ----
  float* wsf   = (float*)d_ws;
  float* xz    = wsf;                                       // 16384*4096
  float* xact  = xz   + (size_t)M_ROWS * XZS;               // 16384*2048
  float* xdbl  = xact + (size_t)M_ROWS * DINNER;            // 16384*96
  float* dt    = xdbl + (size_t)M_ROWS * (DTRANK + 2 * DSTATE);
  float* ycomb = dt   + (size_t)M_ROWS * DINNER;
  bf16_t* wsb     = (bf16_t*)(ycomb + (size_t)M_ROWS * DINNER);
  bf16_t* hbuf_h  = wsb;                                    // 16384*1024
  bf16_t* xact_h  = hbuf_h + (size_t)M_ROWS * DMODEL;       // 16384*2048
  bf16_t* xdbl_h  = xact_h + (size_t)M_ROWS * DINNER;       // 16384*96
  bf16_t* ycomb_h = xdbl_h + (size_t)M_ROWS * 96;           // 16384*2048
  bf16_t* inw_h   = ycomb_h + (size_t)M_ROWS * DINNER;      // 4096*1024
  bf16_t* outw_h  = inw_h  + (size_t)XZS * DMODEL;          // 1024*2048
  bf16_t* xprj_h[2] = { outw_h + (size_t)DMODEL * DINNER, nullptr };
  xprj_h[1] = xprj_h[0] + (size_t)96 * DINNER;
  bf16_t* dtw_h[2] = { xprj_h[1] + (size_t)96 * DINNER, nullptr };
  dtw_h[1] = dtw_h[0] + (size_t)DINNER * DTRANK;

  // ---- weight bf16 casts (cheap, bandwidth-trivial) ----
  cvt_f32_bf16_kernel<<<dim3((XZS * DMODEL / 4) / 256), 256, 0, stream>>>(
      in_w, inw_h, XZS * DMODEL / 4);
  cvt_f32_bf16_kernel<<<dim3((DMODEL * DINNER / 4) / 256), 256, 0, stream>>>(
      out_w, outw_h, DMODEL * DINNER / 4);
  for (int t = 0; t < 2; ++t) {
    cvt_f32_bf16_kernel<<<dim3((96 * DINNER / 4) / 256), 256, 0, stream>>>(
        xprj[t], xprj_h[t], 96 * DINNER / 4);
    cvt_f32_bf16_kernel<<<dim3((DINNER * DTRANK / 4) / 256), 256, 0, stream>>>(
        dtw[t], dtw_h[t], DINNER * DTRANK / 4);
  }

  // 1) residual add + LN + pos  (h emitted as bf16)
  ln_fused_kernel<<<dim3(M_ROWS), dim3(256), 0, stream>>>(
      hidden, resid, pos, nw, nb, resid_out, hbuf_h);

  // 2) in_proj: (16384x1024)bf16 x (4096x1024)^T bf16 -> xz f32
  gemm_wmma_bf16<0><<<dim3(XZS / 128, M_ROWS / 128), dim3(256), 0, stream>>>(
      hbuf_h, DMODEL, inw_h, DMODEL, xz, XZS, XZS, nullptr);

  const int conv_blocks = (int)(((long long)M_ROWS * DINNER) / 256);
  const int scan_blocks = (B_ * DINNER * DSTATE) / 256;
  const int xdbl_c4 = M_ROWS * 96 / 4;

  // 3) forward branch
  conv_silu_kernel<0><<<dim3(conv_blocks), dim3(256), 0, stream>>>(
      xz, conv_w[0], conv_b[0], xact, xact_h);
  gemm_wmma_bf16<0><<<dim3(1, M_ROWS / 128), dim3(256), 0, stream>>>(
      xact_h, DINNER, xprj_h[0], DINNER, xdbl, 96, 96, nullptr);
  cvt_f32_bf16_kernel<<<dim3(xdbl_c4 / 256), 256, 0, stream>>>(xdbl, xdbl_h, xdbl_c4);
  gemm_wmma_bf16<1><<<dim3(DINNER / 128, M_ROWS / 128), dim3(256), 0, stream>>>(
      xdbl_h, 96, dtw_h[0], DTRANK, dt, DINNER, DINNER, dtb[0]);
  scan_kernel<0><<<dim3(scan_blocks), dim3(256), 0, stream>>>(
      xact, dt, xdbl, alog[0], Dvec[0], xz + DINNER, ycomb);

  // 4) backward branch (flipped coordinates, accumulate into ycomb)
  conv_silu_kernel<1><<<dim3(conv_blocks), dim3(256), 0, stream>>>(
      xz, conv_w[1], conv_b[1], xact, xact_h);
  gemm_wmma_bf16<0><<<dim3(1, M_ROWS / 128), dim3(256), 0, stream>>>(
      xact_h, DINNER, xprj_h[1], DINNER, xdbl, 96, 96, nullptr);
  cvt_f32_bf16_kernel<<<dim3(xdbl_c4 / 256), 256, 0, stream>>>(xdbl, xdbl_h, xdbl_c4);
  gemm_wmma_bf16<1><<<dim3(DINNER / 128, M_ROWS / 128), dim3(256), 0, stream>>>(
      xdbl_h, 96, dtw_h[1], DTRANK, dt, DINNER, DINNER, dtb[1]);
  scan_kernel<1><<<dim3(scan_blocks), dim3(256), 0, stream>>>(
      xact, dt, xdbl, alog[1], Dvec[1], xz + DINNER, ycomb);

  // 5) out_proj: ycomb(bf16) x (1024x2048)^T bf16 -> out f32
  cvt_f32_bf16_kernel<<<dim3((M_ROWS * DINNER / 4) / 256), 256, 0, stream>>>(
      ycomb, ycomb_h, M_ROWS * DINNER / 4);
  gemm_wmma_bf16<0><<<dim3(DMODEL / 128, M_ROWS / 128), dim3(256), 0, stream>>>(
      ycomb_h, DINNER, outw_h, DINNER, out, DMODEL, DMODEL, nullptr);
}
